// MoEGate_53541062312117
// MI455X (gfx1250) — matrix operand order repaired and checked by reading, not compiled, not run on hardware
//
#include <hip/hip_runtime.h>

#define HID   2048
#define NEXP  64
#define MT    32       // tokens per block
#define KC    64       // K chunk (elements)
#define NK    (HID / KC)
#define NGRP  8
#define TOPKG 3
#define TOPK  6

typedef _Float16 v16h __attribute__((ext_vector_type(16)));
typedef _Float16 v8h  __attribute__((ext_vector_type(8)));
typedef float    v8f  __attribute__((ext_vector_type(8)));
typedef float    v4f  __attribute__((ext_vector_type(4)));
typedef int      b128v __attribute__((vector_size(16)));   // matches builtin param

struct __align__(16) U4 { unsigned x, y, z, w; };
struct U8 { U4 a, b; };

#if defined(__gfx1250__) && \
    __has_builtin(__builtin_amdgcn_global_load_async_to_lds_b128) && \
    __has_builtin(__builtin_amdgcn_s_wait_asynccnt)
#define USE_ASYNC 1
#else
#define USE_ASYNC 0
#endif

// Build a v16h WMMA fragment from two 16-byte LDS segments (ISA 7.12.2 layout).
__device__ __forceinline__ v16h ld_frag(const _Float16* p0, const _Float16* p1) {
    U8 u;
    u.a = *(const U4*)p0;
    u.b = *(const U4*)p1;
    return __builtin_bit_cast(v16h, u);
}

__global__ void moe_cvt_w_f16(const float* __restrict__ w,
                              _Float16* __restrict__ o, int n) {
    int i = blockIdx.x * blockDim.x + threadIdx.x;
    if (i < n) o[i] = (_Float16)w[i];
}

__global__ __launch_bounds__(256)
void moe_gate_kernel(const float* __restrict__ X,
                     const _Float16* __restrict__ W,
                     int* __restrict__ out_idx,
                     float* __restrict__ out_w) {
    __shared__ _Float16 sX[2][MT][KC];     // 2 x 4 KB  (double buffered)
    __shared__ _Float16 sW[2][NEXP][KC];   // 2 x 8 KB
    __shared__ float    sL[MT][NEXP + 8];  // logits, padded

    const int tid  = threadIdx.x;
    const int lane = tid & 31;
    const int wave = tid >> 5;
    const int mt   = wave >> 2;            // 0..1  (M tile)
    const int nt   = wave & 3;             // 0..3  (N tile)
    const int tok0 = blockIdx.x * MT;

    v8f acc = {};

    // cooperative-load mapping
    const int xrow = tid >> 3;             // 0..31, 8 floats each
    const int xcs  = (tid & 7) * 8;
    const int wrow = tid >> 2;             // 0..63, 16 halves each
    const int wcs  = (tid & 3) * 16;
    const float*    Xg = X + (size_t)(tok0 + xrow) * HID + xcs;
    const _Float16* Wg = W + (size_t)wrow * HID + wcs;

    // A fragment addressing (16-bit A 16x32): lanes<16 hold K{0..7,16..23}
    const int arow = mt * 16 + (lane & 15);
    const int alo  = (lane >= 16) ? 8 : 0;
    // B fragment addressing (16-bit B 32x16): lanes<16 hold K{0..7,16..23}
    const int bcol = nt * 16 + (lane & 15);
    const int blo  = (lane >= 16) ? 8 : 0;

    auto stage = [&](int kc, int buf) {
        // X chunk: fp32 global (streaming, non-temporal) -> cvt -> f16 LDS
        const float* xg = Xg + kc;
        if (kc + KC < HID) __builtin_prefetch(xg + KC, 0, 0);
        v4f f0 = __builtin_nontemporal_load((const v4f*)xg);
        v4f f1 = __builtin_nontemporal_load((const v4f*)(xg + 4));
        v8h hx;
        hx[0] = (_Float16)f0.x; hx[1] = (_Float16)f0.y;
        hx[2] = (_Float16)f0.z; hx[3] = (_Float16)f0.w;
        hx[4] = (_Float16)f1.x; hx[5] = (_Float16)f1.y;
        hx[6] = (_Float16)f1.z; hx[7] = (_Float16)f1.w;
        *(v8h*)&sX[buf][xrow][xcs] = hx;

        // W chunk: fp16 L2 -> LDS (async DMA, tracked by ASYNCcnt)
        const _Float16* wg = Wg + kc;
#if USE_ASYNC
        __builtin_amdgcn_global_load_async_to_lds_b128(
            (__attribute__((address_space(1))) b128v*)wg,
            (__attribute__((address_space(3))) b128v*)&sW[buf][wrow][wcs], 0, 0);
        __builtin_amdgcn_global_load_async_to_lds_b128(
            (__attribute__((address_space(1))) b128v*)(wg + 8),
            (__attribute__((address_space(3))) b128v*)&sW[buf][wrow][wcs + 8], 0, 0);
#else
        *(U4*)&sW[buf][wrow][wcs]     = *(const U4*)wg;
        *(U4*)&sW[buf][wrow][wcs + 8] = *(const U4*)(wg + 8);
#endif
    };

    // prologue: stage chunk 0
    stage(0, 0);
#if USE_ASYNC
    __builtin_amdgcn_s_wait_asynccnt(0);
#endif
    __syncthreads();

    for (int it = 0; it < NK; ++it) {
        const int buf = it & 1;
        if (it + 1 < NK) stage((it + 1) * KC, buf ^ 1);

        #pragma unroll
        for (int s = 0; s < 2; ++s) {
            v16h a = ld_frag(&sX[buf][arow][s * 32 + alo],
                             &sX[buf][arow][s * 32 + alo + 16]);
            v16h b = ld_frag(&sW[buf][bcol][s * 32 + blo],
                             &sW[buf][bcol][s * 32 + blo + 16]);
            acc = __builtin_amdgcn_wmma_f32_16x16x32_f16(
                      false, a, false, b, (short)0, acc, false, false);
        }
#if USE_ASYNC
        __builtin_amdgcn_s_wait_asynccnt(0);
#endif
        __syncthreads();
    }

    // ---- spill C tile to LDS (32x64 logits) ----
    const int crow = mt * 16 + ((lane >= 16) ? 8 : 0);
    const int ccol = nt * 16 + (lane & 15);
    #pragma unroll
    for (int v = 0; v < 8; ++v) sL[crow + v][ccol] = acc[v];
    __syncthreads();

    // ---- phase 2: softmax + group-limited top-k (wave per 4 tokens) ----
    for (int i = 0; i < 4; ++i) {
        const int r  = wave * 4 + i;
        const int gt = tok0 + r;

        float s0 = sL[r][lane];
        float s1 = sL[r][lane + 32];

        float m = fmaxf(s0, s1);
        #pragma unroll
        for (int o = 16; o > 0; o >>= 1) m = fmaxf(m, __shfl_xor(m, o, 32));
        float p0 = __expf(s0 - m), p1 = __expf(s1 - m);
        float sum = p0 + p1;
        #pragma unroll
        for (int o = 16; o > 0; o >>= 1) sum += __shfl_xor(sum, o, 32);
        float inv = 1.0f / sum;
        p0 *= inv; p1 *= inv;

        // group maxima: 8 experts/group -> 8-lane clusters (lo) + (hi)
        float g0 = p0, g1 = p1;
        #pragma unroll
        for (int o = 1; o < 8; o <<= 1) {
            g0 = fmaxf(g0, __shfl_xor(g0, o, 32));
            g1 = fmaxf(g1, __shfl_xor(g1, o, 32));
        }
        float gs[NGRP];
        #pragma unroll
        for (int j = 0; j < 4; ++j) {
            gs[j]     = __shfl(g0, j * 8, 32);
            gs[4 + j] = __shfl(g1, j * 8, 32);
        }
        unsigned gmask = 0;
        #pragma unroll
        for (int k = 0; k < TOPKG; ++k) {
            int bj = 0; float bv = -1.0f;
            #pragma unroll
            for (int j = 0; j < NGRP; ++j)
                if (!((gmask >> j) & 1) && gs[j] > bv) { bv = gs[j]; bj = j; }
            gmask |= 1u << bj;
        }
        float q0 = ((gmask >> (lane >> 3)) & 1)       ? p0 : 0.0f;
        float q1 = ((gmask >> (4 + (lane >> 3))) & 1) ? p1 : 0.0f;

        float tv[TOPK]; int ti[TOPK];
        #pragma unroll
        for (int k = 0; k < TOPK; ++k) {
            float bv; int bi;
            if (q0 >= q1) { bv = q0; bi = lane; } else { bv = q1; bi = lane + 32; }
            #pragma unroll
            for (int o = 16; o > 0; o >>= 1) {
                float ov = __shfl_xor(bv, o, 32);
                int   oj = __shfl_xor(bi, o, 32);
                if (ov > bv || (ov == bv && oj < bi)) { bv = ov; bi = oj; }
            }
            tv[k] = bv; ti[k] = bi;
            if (lane == (bi & 31)) { if (bi < 32) q0 = -1.0f; else q1 = -1.0f; }
        }
        float s6 = 1e-20f;
        #pragma unroll
        for (int k = 0; k < TOPK; ++k) s6 += tv[k];
        float r6 = 1.0f / s6;
        if (lane == 0) {
            #pragma unroll
            for (int k = 0; k < TOPK; ++k) {
                out_idx[(size_t)gt * TOPK + k] = ti[k];
                out_w  [(size_t)gt * TOPK + k] = tv[k] * r6;
            }
        }
    }
}

extern "C" void kernel_launch(void* const* d_in, const int* in_sizes, int n_in,
                              void* d_out, int out_size, void* d_ws, size_t ws_size,
                              hipStream_t stream) {
    const float* x = (const float*)d_in[0];   // (4,4096,2048) fp32
    const float* w = (const float*)d_in[1];   // (64,2048) fp32
    const int T = in_sizes[0] / HID;          // 16384 tokens

    _Float16* wh = (_Float16*)d_ws;           // 256 KB fp16 weights
    const int nW = NEXP * HID;
    moe_cvt_w_f16<<<(nW + 255) / 256, 256, 0, stream>>>(w, wh, nW);

    int*   oi = (int*)d_out;                       // topk_idx  (T,6) int32
    float* ow = (float*)d_out + (size_t)T * TOPK;  // topk_weight (T,6) fp32
    moe_gate_kernel<<<T / MT, 256, 0, stream>>>(x, wh, oi, ow);
}
